// NormalizingFlow2D_86457691669063
// MI455X (gfx1250) — compile-verified, matrix operand order checked by reading, not run on hardware
//
#include <hip/hip_runtime.h>

typedef __attribute__((ext_vector_type(16))) _Float16 v16h;
typedef __attribute__((ext_vector_type(8)))  _Float16 v8h;
typedef __attribute__((ext_vector_type(8)))  float    v8f;

#define ROWS    128      // batch rows per workgroup
#define THREADS 256      // 8 wave32s
#define WIDTH   128      // MLP hidden width
#define STRIDE  136      // padded LDS row stride in halves (272B -> conflict-free b128 column reads)
#define NLAYERS 8
#define BATCH   262144

// ---- CDNA5 async global->LDS copy (ASYNCcnt-tracked, no VGPR round-trip) ----
// Generic pointers to LDS carry the wave-relative LDS byte offset in their low
// 32 bits (ISA 10.2), so pass that as VDST.
__device__ __forceinline__ void async_copy16(unsigned lds_off, const void* gptr) {
  unsigned long long ga = (unsigned long long)(uintptr_t)gptr;
  asm volatile("global_load_async_to_lds_b128 %0, %1, off"
               :: "v"(lds_off), "v"(ga) : "memory");
}
__device__ __forceinline__ void wait_async0() {
  asm volatile("s_wait_asynccnt 0x0" ::: "memory");
}

// Combine two 128-bit LDS loads (K=0..7[+8 hi-lane], K=16..23[+8 hi-lane]) into one
// 16-half WMMA operand, matching the ISA 16-bit 16x32 A/B layouts.
__device__ __forceinline__ v16h ld_tile(const _Float16* p) {
  v8h lo = *(const v8h*)p;          // VGPRs 0-3: K = base + 0..7 (per lane half)
  v8h hi = *(const v8h*)(p + 16);   // VGPRs 4-7: K = base + 16..23 (per lane half)
  return __builtin_shufflevector(lo, hi, 0,1,2,3,4,5,6,7,8,9,10,11,12,13,14,15);
}

#define WMMA16(A, B, C) \
  __builtin_amdgcn_wmma_f32_16x16x32_f16(false, (A), false, (B), (short)0, (C), false, false)

// One (16 rows) x (128 out) x (128 in) GEMM slice per wave:
//   actOut = relu(actIn @ W^T + bias), f16 activations in LDS, fp32 accumulate.
// N-tiles processed in pairs: 16 ds_load_b128 issued up front and two
// independent accumulator chains so consecutive v_wmma ops have no RAW
// dependence (only D->A/B needs hazard NOPs per ISA 7.12.1).
__device__ __forceinline__ void gemm_relu(const _Float16* actIn, const _Float16* wmat,
                                          const float* bias, _Float16* actOut,
                                          int R0, int lane) {
  const int m = lane & 15;    // row within A tile / column within B,C tiles
  const int h = lane >> 4;    // hi-half lane selector
  const _Float16* arow = actIn + (R0 + m) * STRIDE + h * 8;
  v16h A0 = ld_tile(arow);
  v16h A1 = ld_tile(arow + 32);
  v16h A2 = ld_tile(arow + 64);
  v16h A3 = ld_tile(arow + 96);
  const int orow = R0 + h * 8;
#pragma unroll
  for (int nt = 0; nt < 8; nt += 2) {
    const _Float16* brow0 = wmat + (nt * 16 + m) * STRIDE + h * 8;
    const _Float16* brow1 = brow0 + 16 * STRIDE;
    v16h B00 = ld_tile(brow0);
    v16h B01 = ld_tile(brow0 + 32);
    v16h B02 = ld_tile(brow0 + 64);
    v16h B03 = ld_tile(brow0 + 96);
    v16h B10 = ld_tile(brow1);
    v16h B11 = ld_tile(brow1 + 32);
    v16h B12 = ld_tile(brow1 + 64);
    v16h B13 = ld_tile(brow1 + 96);
    v8f acc0 = {};
    v8f acc1 = {};
    acc0 = WMMA16(A0, B00, acc0);  acc1 = WMMA16(A0, B10, acc1);
    acc0 = WMMA16(A1, B01, acc0);  acc1 = WMMA16(A1, B11, acc1);
    acc0 = WMMA16(A2, B02, acc0);  acc1 = WMMA16(A2, B12, acc1);
    acc0 = WMMA16(A3, B03, acc0);  acc1 = WMMA16(A3, B13, acc1);
    const float bb0 = bias[nt * 16 + m];
    const float bb1 = bias[nt * 16 + 16 + m];
#pragma unroll
    for (int j = 0; j < 8; ++j) {
      float v0 = acc0[j] + bb0;
      v0 = v0 > 0.f ? v0 : 0.f;
      actOut[(orow + j) * STRIDE + nt * 16 + m] = (_Float16)v0;
      float v1 = acc1[j] + bb1;
      v1 = v1 > 0.f ? v1 : 0.f;
      actOut[(orow + j) * STRIDE + nt * 16 + 16 + m] = (_Float16)v1;
    }
  }
}

// Prologue: fp32 -> f16 weight repack into workspace, layout [net][layer][W2|W3][128*128].
__global__ __launch_bounds__(256) void convert_weights(
    const float* __restrict__ sW2, const float* __restrict__ sW3,
    const float* __restrict__ tW2, const float* __restrict__ tW3,
    _Float16* __restrict__ whalf) {
  unsigned i = blockIdx.x * 256u + threadIdx.x;   // 0 .. 524287
  unsigned arr   = i >> 17;                        // 0:sW2 1:sW3 2:tW2 3:tW3
  unsigned e     = i & 131071u;
  unsigned layer = e >> 14;
  unsigned r     = e & 16383u;
  const float* src = (arr == 0) ? sW2 : (arr == 1) ? sW3 : (arr == 2) ? tW2 : tW3;
  unsigned net = arr >> 1, which = arr & 1u;
  whalf[(((size_t)net * 8 + layer) * 2 + which) * 16384 + r] = (_Float16)src[e];
}

__global__ __launch_bounds__(THREADS) void flow_kernel(
    const float* __restrict__ x, const float* __restrict__ s_scale,
    const float* __restrict__ sW1, const float* __restrict__ sb1,
    const float* __restrict__ sb2, const float* __restrict__ sb3,
    const float* __restrict__ sW4, const float* __restrict__ sb4,
    const float* __restrict__ tW1, const float* __restrict__ tb1,
    const float* __restrict__ tb2, const float* __restrict__ tb3,
    const float* __restrict__ tW4, const float* __restrict__ tb4,
    const _Float16* __restrict__ whalf,
    float* __restrict__ out, float* __restrict__ scat) {
  extern __shared__ char smem[];
  _Float16* actA = (_Float16*)smem;              // [128][136]
  _Float16* actB = actA + ROWS * STRIDE;         // [128][136]
  _Float16* wb2  = actB + ROWS * STRIDE;         // [128][136]
  _Float16* wb3  = wb2 + WIDTH * STRIDE;         // [128][136]
  float* fb   = (float*)(wb3 + WIDTH * STRIDE);
  float* abuf = fb;                // [128]
  float* bbuf = abuf + ROWS;       // [128]
  float* sres = bbuf + ROWS;       // [128]
  float* tres = sres + ROWS;       // [128]
  float* w1l  = tres + ROWS;       // [128]
  float* b1l  = w1l + WIDTH;       // [128]
  float* b2l  = b1l + WIDTH;       // [128]
  float* b3l  = b2l + WIDTH;       // [128]
  float* w4l  = b3l + WIDTH;       // [128]
  float* b4l  = w4l + WIDTH;       // [1]

  const int tid   = threadIdx.x;
  const int lane  = tid & 31;
  const int R0    = (tid >> 5) * 16;      // 16 rows per wave
  const int gRow0 = blockIdx.x * ROWS;

  if (tid < ROWS) {
    abuf[tid] = x[(size_t)(gRow0 + tid) * 2 + 0];
    bbuf[tid] = x[(size_t)(gRow0 + tid) * 2 + 1];
  }
  __syncthreads();

  for (int layer = 0; layer < NLAYERS; ++layer) {
    const float* zsrc = (layer & 1) ? bbuf : abuf;   // even layers condition on a, odd on b
    for (int net = 0; net < 2; ++net) {              // 0 = s-net, 1 = t-net
      const float* W1g = net ? tW1 : sW1;
      const float* b1g = net ? tb1 : sb1;
      const float* b2g = net ? tb2 : sb2;
      const float* b3g = net ? tb3 : sb3;
      const float* W4g = net ? tW4 : sW4;
      const float* b4g = net ? tb4 : sb4;
      const _Float16* w2src = whalf + ((size_t)(net * 8 + layer) * 2 + 0) * (WIDTH * WIDTH);
      const _Float16* w3src = whalf + ((size_t)(net * 8 + layer) * 2 + 1) * (WIDTH * WIDTH);

      // Stage W2/W3 (f16) into padded LDS via async global->LDS B128 DMA:
      // 128 rows x 16 chunks of 8 halves each, per matrix.
      for (int c = tid; c < WIDTH * 16; c += THREADS) {
        int row = c >> 4, col = (c & 15) * 8;
        async_copy16((unsigned)(uintptr_t)(wb2 + row * STRIDE + col),
                     w2src + row * WIDTH + col);
        async_copy16((unsigned)(uintptr_t)(wb3 + row * STRIDE + col),
                     w3src + row * WIDTH + col);
      }
      if (tid < WIDTH) {
        w1l[tid] = W1g[layer * WIDTH + tid];
        b1l[tid] = b1g[layer * WIDTH + tid];
        b2l[tid] = b2g[layer * WIDTH + tid];
        b3l[tid] = b3g[layer * WIDTH + tid];
        w4l[tid] = W4g[layer * WIDTH + tid];
      }
      if (tid == 0) b4l[0] = b4g[layer];
      wait_async0();               // ASYNCcnt == 0: weight tiles landed in LDS
      __syncthreads();

      // Layer 1: h1 = relu(z * W1 + b1)  (outer-product, fp32 VALU, b128 stores)
      for (int e = tid; e < ROWS * 16; e += THREADS) {
        int row = e >> 4, c8 = (e & 15) * 8;
        float z = zsrc[row];
        v8h hv;
#pragma unroll
        for (int j = 0; j < 8; ++j) {
          float v = z * w1l[c8 + j] + b1l[c8 + j];
          hv[j] = (_Float16)(v > 0.f ? v : 0.f);
        }
        *(v8h*)(actA + row * STRIDE + c8) = hv;
      }
      __syncthreads();

      // Layers 2 & 3: 128x128 WMMA GEMMs with fused bias+ReLU.
      gemm_relu(actA, wb2, b2l, actB, R0, lane);
      __syncthreads();
      gemm_relu(actB, wb3, b3l, actA, R0, lane);
      __syncthreads();

      // Layer 4: y = tanh(h3 . W4 + b4)  (fp32 VALU dot per row, b128 reads)
      if (tid < ROWS) {
        const v8h* hrow = (const v8h*)(actA + tid * STRIDE);   // 272B row, 16B aligned
        float acc = b4l[0];
#pragma unroll
        for (int k8 = 0; k8 < 16; ++k8) {
          v8h hv = hrow[k8];
#pragma unroll
          for (int j = 0; j < 8; ++j) acc += (float)hv[j] * w4l[k8 * 8 + j];
        }
        acc = fminf(fmaxf(acc, -15.f), 15.f);
        float e2 = __expf(2.f * acc);
        float y = (e2 - 1.f) / (e2 + 1.f);
        (net ? tres : sres)[tid] = y;
      }
      __syncthreads();
    }

    // Coupling update + log-scale output.
    if (tid < ROWS) {
      float sv = s_scale[layer] * sres[tid];
      float ev = __expf(sv);
      if ((layer & 1) == 0) bbuf[tid] = ev * bbuf[tid] + tres[tid];
      else                  abuf[tid] = ev * abuf[tid] + tres[tid];
      scat[(size_t)layer * BATCH + gRow0 + tid] = sv;
    }
    __syncthreads();

    // Warm next layer's s-net W2 in L2 (global_prefetch_b8).
    if (layer + 1 < NLAYERS) {
      const _Float16* nxt = whalf + (size_t)((layer + 1) * 2) * (WIDTH * WIDTH);
      __builtin_prefetch(nxt + tid * 64, 0, 1);
    }
  }

  if (tid < ROWS) {
    out[(size_t)(gRow0 + tid) * 2 + 0] = abuf[tid];
    out[(size_t)(gRow0 + tid) * 2 + 1] = bbuf[tid];
  }
}

extern "C" void kernel_launch(void* const* d_in, const int* in_sizes, int n_in,
                              void* d_out, int out_size, void* d_ws, size_t ws_size,
                              hipStream_t stream) {
  const float* x       = (const float*)d_in[0];
  const float* s_scale = (const float*)d_in[1];
  const float* sW1 = (const float*)d_in[2],  *sb1 = (const float*)d_in[3];
  const float* sW2 = (const float*)d_in[4],  *sb2 = (const float*)d_in[5];
  const float* sW3 = (const float*)d_in[6],  *sb3 = (const float*)d_in[7];
  const float* sW4 = (const float*)d_in[8],  *sb4 = (const float*)d_in[9];
  const float* tW1 = (const float*)d_in[10], *tb1 = (const float*)d_in[11];
  const float* tW2 = (const float*)d_in[12], *tb2 = (const float*)d_in[13];
  const float* tW3 = (const float*)d_in[14], *tb3 = (const float*)d_in[15];
  const float* tW4 = (const float*)d_in[16], *tb4 = (const float*)d_in[17];

  float* out  = (float*)d_out;
  float* scat = out + (size_t)2 * BATCH;        // (out[B,2], s_cat[L*B]) flat
  _Float16* whalf = (_Float16*)d_ws;            // 1 MB of f16 weights

  convert_weights<<<2048, 256, 0, stream>>>(sW2, sW3, tW2, tW3, whalf);

  const size_t smem =
      (size_t)4 * WIDTH * STRIDE * sizeof(_Float16) +   // actA, actB, wb2, wb3
      (size_t)(4 * ROWS + 5 * WIDTH + 4) * sizeof(float);
  hipFuncSetAttribute((const void*)flow_kernel,
                      hipFuncAttributeMaxDynamicSharedMemorySize, (int)smem);

  flow_kernel<<<BATCH / ROWS, THREADS, smem, stream>>>(
      x, s_scale, sW1, sb1, sb2, sb3, sW4, sb4,
      tW1, tb1, tb2, tb3, tW4, tb4, whalf, out, scat);
}
